// Sophie_24524263260711
// MI455X (gfx1250) — matrix operand order repaired and checked by reading, not compiled, not run on hardware
//
#include <hip/hip_runtime.h>
#include <hip/hip_bf16.h>

// Problem constants (from reference)
#define HD    512      // hidden
#define G4    2048     // 4*HD (gates)
#define NO    4096     // n_others
#define TT    64       // T = OBS + PRED
#define OBSL  32
#define MAPHW 256

typedef __attribute__((ext_vector_type(16))) _Float16 v16h;
typedef __attribute__((ext_vector_type(8)))  float    v8f;

struct U32x8 { uint4 a, b; };

// ---- WMMA fragment loaders (wave32, 16x16x32 f16, per CDNA5 ISA layouts) ----
// A (16x32, MxK) f16, row-major [lda=K-stride in elems]:
//   lane L: row = m0 + (L&15); halves 0..7 = K kb..kb+7, halves 8..15 = K kb+16..kb+23,
//   kb = k0 + (L>=16 ? 8 : 0)
__device__ __forceinline__ v16h load_frag_a(const _Float16* __restrict__ A, int lda,
                                            int m0, int k0, int lane) {
    int row = m0 + (lane & 15);
    int kb  = k0 + ((lane >> 4) << 3);
    const _Float16* p = A + (size_t)row * lda + kb;
    U32x8 r;
    r.a = *reinterpret_cast<const uint4*>(p);
    r.b = *reinterpret_cast<const uint4*>(p + 16);
    return __builtin_bit_cast(v16h, r);
}

// B (32x16, KxN) f16, supplied as B^T row-major (i.e. weight rows = output cols):
//   lane L: col = n0 + (L&15); halves 0..7 = K kb..kb+7, halves 8..15 = K kb+16..kb+23, kb = k0 + (L>=16 ? 8 : 0)
__device__ __forceinline__ v16h load_frag_bt(const _Float16* __restrict__ Bt, int ldb,
                                             int n0, int k0, int lane) {
    int col = n0 + (lane & 15);
    int kb  = k0 + ((lane >> 4) << 3);
    const _Float16* p = Bt + (size_t)col * ldb + kb;
    U32x8 r;
    r.a = *reinterpret_cast<const uint4*>(p);
    r.b = *reinterpret_cast<const uint4*>(p + 16);
    return __builtin_bit_cast(v16h, r);
}

__device__ __forceinline__ v8f wmma_16x16x32(v16h a, v16h b, v8f c) {
    return __builtin_amdgcn_wmma_f32_16x16x32_f16(false, a, false, b, (short)0, c,
                                                  false, false);
}

__device__ __forceinline__ float sigm(float x) { return 1.0f / (1.0f + __expf(-x)); }

// ------------------------- setup kernels -------------------------

__global__ void init_kernel(float* __restrict__ co, _Float16* __restrict__ hoA,
                            _Float16* __restrict__ hoB, float* __restrict__ ht,
                            float* __restrict__ ct, float* __restrict__ mean) {
    int i = blockIdx.x * blockDim.x + threadIdx.x;           // 8192*256 == NO*HD
    co[i]  = 0.0f;
    hoA[i] = (_Float16)0.0f;
    hoB[i] = (_Float16)0.0f;
    if (i < HD) { ht[i] = 0.0f; ct[i] = 0.0f; }
    if (i < 32) mean[i] = 0.0f;
}

__global__ void convert_kernel(const float* __restrict__ whh, const float* __restrict__ wk,
                               const float* __restrict__ wv, _Float16* __restrict__ whh16,
                               _Float16* __restrict__ wkv16) {
    int i = blockIdx.x * blockDim.x + threadIdx.x;
    const int n1 = G4 * HD;          // w_hh elements
    const int n2 = 2 * HD * HD;      // wk|wv elements
    if (i < n1) {
        whh16[i] = (_Float16)whh[i];
    } else if (i < n1 + n2) {
        int j = i - n1;
        int n = j / HD, k = j % HD;
        float w = (n < HD) ? wk[(size_t)n * HD + k] : wv[(size_t)(n - HD) * HD + k];
        wkv16[j] = (_Float16)w;
    }
}

// embedding lookup + conv1(4->16, 3x3 SAME) + relu
__global__ void scene_conv1_kernel(const int* __restrict__ smap, const float* __restrict__ emb,
                                   const float* __restrict__ w1, const float* __restrict__ b1,
                                   float* __restrict__ out1) {
    int i = blockIdx.x, j = threadIdx.x;
    float xin[9][4];
    #pragma unroll
    for (int di = 0; di < 3; ++di) {
        #pragma unroll
        for (int dj = 0; dj < 3; ++dj) {
            int ii = i + di - 1, jj = j + dj - 1;
            float* x = xin[di * 3 + dj];
            if (ii >= 0 && ii < MAPHW && jj >= 0 && jj < MAPHW) {
                const float* e = emb + (size_t)smap[ii * MAPHW + jj] * 4;
                x[0] = e[0]; x[1] = e[1]; x[2] = e[2]; x[3] = e[3];
            } else {
                x[0] = x[1] = x[2] = x[3] = 0.0f;
            }
        }
    }
    for (int oc = 0; oc < 16; ++oc) {
        float acc = b1[oc];
        #pragma unroll
        for (int ic = 0; ic < 4; ++ic)
            #pragma unroll
            for (int p = 0; p < 9; ++p)
                acc += w1[(oc * 4 + ic) * 9 + p] * xin[p][ic];
        out1[(size_t)oc * MAPHW * MAPHW + i * MAPHW + j] = fmaxf(acc, 0.0f);
    }
}

// conv2(16->32, 3x3 SAME) + relu + global-mean partials (atomicAdd per block)
__global__ void scene_conv2_kernel(const float* __restrict__ out1, const float* __restrict__ w2,
                                   const float* __restrict__ b2, float* __restrict__ mean) {
    int i = blockIdx.x, j = threadIdx.x;
    float acc[32];
    for (int oc = 0; oc < 32; ++oc) acc[oc] = b2[oc];
    for (int ic = 0; ic < 16; ++ic) {
        float n9[9];
        #pragma unroll
        for (int di = 0; di < 3; ++di)
            #pragma unroll
            for (int dj = 0; dj < 3; ++dj) {
                int ii = i + di - 1, jj = j + dj - 1;
                n9[di * 3 + dj] = (ii >= 0 && ii < MAPHW && jj >= 0 && jj < MAPHW)
                    ? out1[(size_t)ic * MAPHW * MAPHW + ii * MAPHW + jj] : 0.0f;
            }
        for (int oc = 0; oc < 32; ++oc) {
            const float* w = w2 + (oc * 16 + ic) * 9;
            float s = 0.0f;
            #pragma unroll
            for (int p = 0; p < 9; ++p) s += w[p] * n9[p];
            acc[oc] += s;
        }
    }
    __shared__ float red[256];
    for (int oc = 0; oc < 32; ++oc) {
        red[j] = fmaxf(acc[oc], 0.0f);
        __syncthreads();
        for (int s = 128; s > 0; s >>= 1) {
            if (j < s) red[j] += red[j + s];
            __syncthreads();
        }
        if (j == 0) atomicAdd(mean + oc, red[0]);
        __syncthreads();
    }
}

__global__ void scene_fc_kernel(const float* __restrict__ mean, const float* __restrict__ fcw,
                                const float* __restrict__ fcb, float* __restrict__ scene) {
    int t = threadIdx.x;  // 512
    float acc = fcb[t];
    #pragma unroll
    for (int c = 0; c < 32; ++c)
        acc += (mean[c] * (1.0f / 65536.0f)) * fcw[t * 32 + c];
    scene[t] = acc;
}

// ------------------------- per-timestep kernels -------------------------

// Target LSTM (batch 1): one block of 512 threads; thread t owns hidden unit t.
__global__ void target_lstm_kernel(const float* __restrict__ xt, const float* __restrict__ wih,
                                   const float* __restrict__ whh, const float* __restrict__ bih,
                                   const float* __restrict__ bhh, float* __restrict__ ht,
                                   float* __restrict__ ct) {
    __shared__ float hs[HD];
    int t = threadIdx.x;
    hs[t] = ht[t];
    __syncthreads();
    float x0 = xt[0], x1 = xt[1];
    float g[4];
    #pragma unroll
    for (int gi = 0; gi < 4; ++gi) {
        int n = gi * HD + t;
        float acc = bih[n] + bhh[n] + x0 * wih[2 * n] + x1 * wih[2 * n + 1];
        const float* w = whh + (size_t)n * HD;
        for (int k = 0; k < HD; ++k) acc += hs[k] * w[k];
        g[gi] = acc;
    }
    float cn = sigm(g[1]) * ct[t] + sigm(g[0]) * tanhf(g[2]);
    ct[t] = cn;
    ht[t] = sigm(g[3]) * tanhf(cn);
}

// Others LSTM, fully fused: WMMA GEMM (4096x2048x512) + input/bias + pointwise LSTM.
// Each wave computes a 32(M)x16(h) tile for ALL FOUR gate blocks (n0, n0+512,
// n0+1024, n0+1536) -> 8 accumulators; i/f/g/o line up elementwise per lane.
// Ping-pong (unroll-by-2) register double-buffering: two structurally distinct
// fragment sets alternate, so no inter-buffer register copies are ever emitted
// and loads for step k+1 are always in flight during step k's 8 WMMAs.
__global__ void __launch_bounds__(128)
others_gates_kernel(const _Float16* __restrict__ hoR, const _Float16* __restrict__ whh16,
                    const float* __restrict__ xo, const float* __restrict__ wih,
                    const float* __restrict__ bih, const float* __restrict__ bhh,
                    float* __restrict__ co, _Float16* __restrict__ hoW) {
    const int lane = threadIdx.x & 31;
    const int wave = threadIdx.x >> 5;
    const int n0 = blockIdx.x * 16;                   // hidden tile within each gate
    const int m0 = (blockIdx.y * 4 + wave) * 32;      // 32-row tile
    v8f acc[8] = {};                                  // [mt*4+g]
    // Set C: k-step 0
    v16h a0c = load_frag_a(hoR, HD, m0,      0, lane);
    v16h a1c = load_frag_a(hoR, HD, m0 + 16, 0, lane);
    v16h bc[4];
    #pragma unroll
    for (int g = 0; g < 4; ++g) bc[g] = load_frag_bt(whh16, HD, g * HD + n0, 0, lane);

    for (int kk = 0; kk < HD / 32; kk += 2) {
        const int k1 = (kk + 1) * 32;                               // always < HD
        const int k2 = (kk + 2 < HD / 32) ? (kk + 2) * 32 : k1;     // clamp on last
        __builtin_prefetch(whh16 + (size_t)n0 * HD + k2, 0, 3);
        // Load set N (k+1) while computing with set C (k)
        v16h a0n = load_frag_a(hoR, HD, m0,      k1, lane);
        v16h a1n = load_frag_a(hoR, HD, m0 + 16, k1, lane);
        v16h bn[4];
        #pragma unroll
        for (int g = 0; g < 4; ++g) bn[g] = load_frag_bt(whh16, HD, g * HD + n0, k1, lane);
        #pragma unroll
        for (int g = 0; g < 4; ++g) {
            acc[g]     = wmma_16x16x32(a0c, bc[g], acc[g]);
            acc[4 + g] = wmma_16x16x32(a1c, bc[g], acc[4 + g]);
        }
        // Reload set C (k+2) while computing with set N (k+1)
        a0c = load_frag_a(hoR, HD, m0,      k2, lane);
        a1c = load_frag_a(hoR, HD, m0 + 16, k2, lane);
        #pragma unroll
        for (int g = 0; g < 4; ++g) bc[g] = load_frag_bt(whh16, HD, g * HD + n0, k2, lane);
        #pragma unroll
        for (int g = 0; g < 4; ++g) {
            acc[g]     = wmma_16x16x32(a0n, bn[g], acc[g]);
            acc[4 + g] = wmma_16x16x32(a1n, bn[g], acc[4 + g]);
        }
    }

    // Epilogue: D layout -> lane owns col hcol, rows mbase..mbase+7 per m-subtile
    const int hcol = n0 + (lane & 15);
    const float bs0 = bih[hcol] + bhh[hcol];
    const float bs1 = bih[HD + hcol] + bhh[HD + hcol];
    const float bs2 = bih[2 * HD + hcol] + bhh[2 * HD + hcol];
    const float bs3 = bih[3 * HD + hcol] + bhh[3 * HD + hcol];
    const float wi00 = wih[2 * hcol],            wi01 = wih[2 * hcol + 1];
    const float wi10 = wih[2 * (HD + hcol)],     wi11 = wih[2 * (HD + hcol) + 1];
    const float wi20 = wih[2 * (2 * HD + hcol)], wi21 = wih[2 * (2 * HD + hcol) + 1];
    const float wi30 = wih[2 * (3 * HD + hcol)], wi31 = wih[2 * (3 * HD + hcol) + 1];
    #pragma unroll
    for (int mt = 0; mt < 2; ++mt) {
        const int mbase = m0 + mt * 16 + ((lane >> 4) << 3);
        #pragma unroll
        for (int r = 0; r < 8; ++r) {
            int m = mbase + r;
            float x0 = xo[2 * m], x1 = xo[2 * m + 1];
            float gi = acc[mt * 4 + 0][r] + bs0 + x0 * wi00 + x1 * wi01;
            float gf = acc[mt * 4 + 1][r] + bs1 + x0 * wi10 + x1 * wi11;
            float gg = acc[mt * 4 + 2][r] + bs2 + x0 * wi20 + x1 * wi21;
            float go = acc[mt * 4 + 3][r] + bs3 + x0 * wi30 + x1 * wi31;
            size_t idx = (size_t)m * HD + hcol;
            float cn = sigm(gf) * co[idx] + sigm(gi) * tanhf(gg);
            co[idx] = cn;
            hoW[idx] = (_Float16)(sigm(go) * tanhf(cn));
        }
    }
}

// K|V projection: (4096x512) @ (512x1024) with bias, WMMA, f32 output [NO x 2*HD].
// Wave computes 32(M) x 64(N); same ping-pong double-buffering as gates kernel.
__global__ void __launch_bounds__(128)
kv_gemm_kernel(const _Float16* __restrict__ ho, const _Float16* __restrict__ wkv16,
               const float* __restrict__ bk, const float* __restrict__ bv,
               float* __restrict__ kv) {
    const int lane = threadIdx.x & 31;
    const int wave = threadIdx.x >> 5;
    const int nb = blockIdx.x * 64;
    const int m0 = (blockIdx.y * 4 + wave) * 32;
    v8f acc[8] = {};                                  // [mt*4+s]
    v16h a0c = load_frag_a(ho, HD, m0,      0, lane);
    v16h a1c = load_frag_a(ho, HD, m0 + 16, 0, lane);
    v16h bc[4];
    #pragma unroll
    for (int s = 0; s < 4; ++s) bc[s] = load_frag_bt(wkv16, HD, nb + s * 16, 0, lane);

    for (int kk = 0; kk < HD / 32; kk += 2) {
        const int k1 = (kk + 1) * 32;
        const int k2 = (kk + 2 < HD / 32) ? (kk + 2) * 32 : k1;
        __builtin_prefetch(wkv16 + (size_t)nb * HD + k2, 0, 3);
        v16h a0n = load_frag_a(ho, HD, m0,      k1, lane);
        v16h a1n = load_frag_a(ho, HD, m0 + 16, k1, lane);
        v16h bn[4];
        #pragma unroll
        for (int s = 0; s < 4; ++s) bn[s] = load_frag_bt(wkv16, HD, nb + s * 16, k1, lane);
        #pragma unroll
        for (int s = 0; s < 4; ++s) {
            acc[s]     = wmma_16x16x32(a0c, bc[s], acc[s]);
            acc[4 + s] = wmma_16x16x32(a1c, bc[s], acc[4 + s]);
        }
        a0c = load_frag_a(ho, HD, m0,      k2, lane);
        a1c = load_frag_a(ho, HD, m0 + 16, k2, lane);
        #pragma unroll
        for (int s = 0; s < 4; ++s) bc[s] = load_frag_bt(wkv16, HD, nb + s * 16, k2, lane);
        #pragma unroll
        for (int s = 0; s < 4; ++s) {
            acc[s]     = wmma_16x16x32(a0n, bn[s], acc[s]);
            acc[4 + s] = wmma_16x16x32(a1n, bn[s], acc[4 + s]);
        }
    }

    #pragma unroll
    for (int mt = 0; mt < 2; ++mt) {
        const int mbase = m0 + mt * 16 + ((lane >> 4) << 3);
        #pragma unroll
        for (int s = 0; s < 4; ++s) {
            int n = nb + s * 16 + (lane & 15);
            float bias = (n < HD) ? bk[n] : bv[n - HD];
            #pragma unroll
            for (int r = 0; r < 8; ++r) {
                kv[(size_t)(mbase + r) * (2 * HD) + n] = acc[mt * 4 + s][r] + bias;
            }
        }
    }
}

// Fused q-projection + attention scores + softmax + social + output head.
__global__ void __launch_bounds__(1024)
attn_out_kernel(const float* __restrict__ ht, const float* __restrict__ kv,
                const float* __restrict__ scene, const float* __restrict__ wq,
                const float* __restrict__ bq, const float* __restrict__ outw,
                const float* __restrict__ outb, float* __restrict__ outp) {
    __shared__ float hs[HD];
    __shared__ float qs[HD];
    __shared__ float ss[HD];
    __shared__ float sc[NO];
    __shared__ float red[1024];
    const int t = threadIdx.x;
    if (t < HD) hs[t] = ht[t];
    __syncthreads();
    if (t < HD) {  // q = ht @ wq^T + bq
        float acc = bq[t];
        const float* w = wq + (size_t)t * HD;
        for (int k = 0; k < HD; ++k) acc += hs[k] * w[k];
        qs[t] = acc;
    }
    __syncthreads();
    const float invT = 0.0441941738241592f;  // 1/sqrt(512)
    float s_[4];
    float lmax = -3.4e38f;
    #pragma unroll
    for (int p = 0; p < 4; ++p) {  // scores: 4 rows per thread
        int j = t + p * 1024;
        const float* kp = kv + (size_t)j * (2 * HD);
        float acc = 0.0f;
        for (int k = 0; k < HD; ++k) acc += qs[k] * kp[k];
        acc *= invT;
        s_[p] = acc;
        lmax = fmaxf(lmax, acc);
    }
    red[t] = lmax; __syncthreads();
    for (int s = 512; s > 0; s >>= 1) {
        if (t < s) red[t] = fmaxf(red[t], red[t + s]);
        __syncthreads();
    }
    float mx = red[0];
    __syncthreads();
    float lsum = 0.0f;
    #pragma unroll
    for (int p = 0; p < 4; ++p) {
        float e = __expf(s_[p] - mx);
        sc[t + p * 1024] = e;
        lsum += e;
    }
    red[t] = lsum; __syncthreads();
    for (int s = 512; s > 0; s >>= 1) {
        if (t < s) red[t] += red[t + s];
        __syncthreads();
    }
    float inv = 1.0f / red[0];
    __syncthreads();
    if (t < HD) {  // social = attn @ v  (coalesced column reads of V half)
        float acc = 0.0f;
        const float* vcol = kv + HD + t;
        for (int j = 0; j < NO; ++j) acc += sc[j] * vcol[(size_t)j * (2 * HD)];
        ss[t] = acc * inv;
    }
    __syncthreads();
    float p0 = 0.0f, p1 = 0.0f;
    if (t < HD) {
        float s = hs[t] + ss[t] + scene[t];
        p0 = s * outw[t];
        p1 = s * outw[HD + t];
    }
    red[t] = p0; __syncthreads();
    for (int s = 512; s > 0; s >>= 1) { if (t < s) red[t] += red[t + s]; __syncthreads(); }
    if (t == 0) outp[0] = red[0] + outb[0];
    __syncthreads();
    red[t] = p1; __syncthreads();
    for (int s = 512; s > 0; s >>= 1) { if (t < s) red[t] += red[t + s]; __syncthreads(); }
    if (t == 0) outp[1] = red[0] + outb[1];
}

// ------------------------- host launcher -------------------------

extern "C" void kernel_launch(void* const* d_in, const int* in_sizes, int n_in,
                              void* d_out, int out_size, void* d_ws, size_t ws_size,
                              hipStream_t stream) {
    (void)in_sizes; (void)n_in; (void)out_size; (void)ws_size;
    const float* target    = (const float*)d_in[0];
    const float* others    = (const float*)d_in[1];
    const int*   scene_map = (const int*)  d_in[2];
    const float* emb       = (const float*)d_in[3];
    const float* c1w       = (const float*)d_in[4];
    const float* c1b       = (const float*)d_in[5];
    const float* c2w       = (const float*)d_in[6];
    const float* c2b       = (const float*)d_in[7];
    const float* fcw       = (const float*)d_in[8];
    const float* fcb       = (const float*)d_in[9];
    const float* wih       = (const float*)d_in[10];
    const float* whh       = (const float*)d_in[11];
    const float* bih       = (const float*)d_in[12];
    const float* bhh       = (const float*)d_in[13];
    const float* wq        = (const float*)d_in[14];
    const float* bq        = (const float*)d_in[15];
    const float* wk        = (const float*)d_in[16];
    const float* bk        = (const float*)d_in[17];
    const float* wv        = (const float*)d_in[18];
    const float* bv        = (const float*)d_in[19];
    const float* outw      = (const float*)d_in[20];
    const float* outb      = (const float*)d_in[21];

    char* ws = (char*)d_ws;
    size_t off = 0;
    auto take = [&](size_t bytes) -> char* {
        char* p = ws + off;
        off = (off + bytes + 255) & ~(size_t)255;
        return p;
    };
    _Float16* whh16 = (_Float16*)take((size_t)G4 * HD * 2);       // 2 MB
    _Float16* wkv16 = (_Float16*)take((size_t)2 * HD * HD * 2);   // 1 MB
    _Float16* hoA   = (_Float16*)take((size_t)NO * HD * 2);       // 4 MB
    _Float16* hoB   = (_Float16*)take((size_t)NO * HD * 2);       // 4 MB
    float*    co    = (float*)   take((size_t)NO * HD * 4);       // 8 MB
    float*    kv    = (float*)   take((size_t)NO * 2 * HD * 4);   // 16 MB
    float*    ht    = (float*)   take(HD * 4);
    float*    ct    = (float*)   take(HD * 4);
    float*    scene = (float*)   take(HD * 4);
    float*    mean  = (float*)   take(32 * 4);
    float*    conv1 = (float*)   take((size_t)16 * MAPHW * MAPHW * 4);  // 4 MB

    // One-time setup (re-run every call: deterministic, stateless across calls)
    init_kernel<<<(NO * HD) / 256, 256, 0, stream>>>(co, hoA, hoB, ht, ct, mean);
    convert_kernel<<<(G4 * HD + 2 * HD * HD + 255) / 256, 256, 0, stream>>>(
        whh, wk, wv, whh16, wkv16);
    scene_conv1_kernel<<<MAPHW, MAPHW, 0, stream>>>(scene_map, emb, c1w, c1b, conv1);
    scene_conv2_kernel<<<MAPHW, MAPHW, 0, stream>>>(conv1, c2w, c2b, mean);
    scene_fc_kernel<<<1, HD, 0, stream>>>(mean, fcw, fcb, scene);

    float* outp = (float*)d_out;
    for (int t = 0; t < TT; ++t) {
        const float* xt = target + (size_t)t * 2;
        int ti = (t < OBSL) ? t : (OBSL - 1);
        const float* xo = others + (size_t)ti * NO * 2;
        const _Float16* hoR = (t & 1) ? hoB : hoA;   // double buffer (WAR across blocks)
        _Float16*       hoW = (t & 1) ? hoA : hoB;

        target_lstm_kernel<<<1, HD, 0, stream>>>(xt, wih, whh, bih, bhh, ht, ct);
        others_gates_kernel<<<dim3(HD / 16, NO / 128), 128, 0, stream>>>(
            hoR, whh16, xo, wih, bih, bhh, co, hoW);
        kv_gemm_kernel<<<dim3(2 * HD / 64, NO / 128), 128, 0, stream>>>(
            hoW, wkv16, bk, bv, kv);
        attn_out_kernel<<<1, 1024, 0, stream>>>(
            ht, kv, scene, wq, bq, outw, outb, outp + (size_t)t * 2);
    }
}
